// GNNs_75763223102156
// MI455X (gfx1250) — matrix operand order repaired and checked by reading, not compiled, not run on hardware
//
#include <hip/hip_runtime.h>
#include <math.h>

typedef __bf16 bf16_t;
typedef __attribute__((ext_vector_type(16))) __bf16 v16bf;
typedef __attribute__((ext_vector_type(8)))  float  v8f;

#define DGR 128   // d_graph
#define HID 256   // hidden
#define NGRAPH 256
#define NOUT 2

// ---------------------------------------------------------------- utilities
__global__ void gnn_zero_f32(float* __restrict__ p, size_t n) {
  size_t i = (size_t)blockIdx.x * blockDim.x + threadIdx.x;
  size_t stride = (size_t)gridDim.x * blockDim.x;
  for (; i < n; i += stride) p[i] = 0.0f;
}

// Pack f32 weight W[K][Nw] into bf16 WMMA B-fragment order:
// out[((tile*(K/32) + kkidx)*32 + lane)*16 + e] = W[kkidx*32 + (e<8?0:16) + (lane<16?0:8) + (e&7)][tile*16 + (lane&15)]
__global__ void gnn_pack_w(const float* __restrict__ W, bf16_t* __restrict__ out, int K, int Nw) {
  int i = blockIdx.x * blockDim.x + threadIdx.x;
  if (i >= K * Nw) return;
  int e     = i & 15;
  int lane  = (i >> 4) & 31;
  int nkk   = K >> 5;
  int kkidx = (i >> 9) % nkk;
  int tile  = (i >> 9) / nkk;
  int col   = tile * 16 + (lane & 15);
  int k     = kkidx * 32 + ((e < 8) ? 0 : 16) + ((lane < 16) ? 0 : 8) + (e & 7);
  out[i] = (bf16_t)W[k * Nw + col];
}

// ---------------------------------------------------------------- embedding gather
__global__ void gnn_gather_embed(const int* __restrict__ feat,
                                 const float* __restrict__ emb,
                                 float* __restrict__ h, int N) {
  int tid = blockIdx.x * blockDim.x + threadIdx.x;   // N*32 threads, float4 each
  if (tid >= N * 32) return;
  int node = tid >> 5, c4 = tid & 31;
  const float4* e4 = (const float4*)(emb + (size_t)feat[node] * DGR);
  ((float4*)(h + (size_t)node * DGR))[c4] = e4[c4];
}

// ---------------------------------------------------------------- edge phase
__global__ void gnn_deg_count(const int* __restrict__ dst, float* __restrict__ deg, int E) {
  int i = blockIdx.x * blockDim.x + threadIdx.x;
  if (i < E) atomicAdd(deg + dst[i], 1.0f);
}

__global__ void gnn_edge_scatter(const int* __restrict__ src, const int* __restrict__ dst,
                                 const float* __restrict__ h, float* __restrict__ agg, int E) {
  int tid = blockIdx.x * blockDim.x + threadIdx.x;   // E*32 threads, wave per edge
  if (tid >= E * 32) return;
  int e = tid >> 5, lane = tid & 31;
  int s = src[e], d = dst[e];
  float4 v = ((const float4*)(h + (size_t)s * DGR))[lane];
  float* ap = agg + (size_t)d * DGR + lane * 4;
  atomicAdd(ap + 0, v.x); atomicAdd(ap + 1, v.y);
  atomicAdd(ap + 2, v.z); atomicAdd(ap + 3, v.w);
}

// ---------------------------------------------------------------- fused SAGE node update
// hout = relu(h @ Ws + (agg/max(deg,1)) @ Wn + b)
// block = 32 rows x 128 cols; 8 waves, each wave: 1 col tile x 2 row tiles,
// B fragments (packed, L2-resident) reused across both row tiles.
__global__ __launch_bounds__(256)
void gnn_sage_update(const float* __restrict__ hin, const float* __restrict__ agg,
                     const float* __restrict__ deg,
                     const bf16_t* __restrict__ WselfPk, const bf16_t* __restrict__ WneighPk,
                     const float* __restrict__ bias, float* __restrict__ hout, int N) {
  __shared__ bf16_t As[32][136];   // self features, bf16
  __shared__ bf16_t An[32][136];   // mean-aggregated neighbor features, bf16
  __shared__ float  sInv[32];

  const int rowBase = blockIdx.x * 32;
  const int t = threadIdx.x;

  if (t < 32) sInv[t] = 1.0f / fmaxf(deg[rowBase + t], 1.0f);
  __syncthreads();

  {  // stage 32x128 of each A matrix: 16 contiguous elems per thread
    int idx = t * 16;
    int r = idx >> 7, c = idx & 127;
    const float* hp = hin + (size_t)(rowBase + r) * DGR + c;
    const float* ap = agg + (size_t)(rowBase + r) * DGR + c;
    float inv = sInv[r];
#pragma unroll
    for (int i = 0; i < 16; ++i) {
      As[r][c + i] = (bf16_t)hp[i];
      An[r][c + i] = (bf16_t)(ap[i] * inv);
    }
  }
  __syncthreads();

  const int wave = t >> 5;              // 8 waves -> 8 column tiles of 16
  const int lane = t & 31;
  const int mrow = lane & 15;
  const int kbA  = (lane >> 4) * 8;     // A-fragment K sub-offset per ISA layout
  const int wcol = wave * 16 + mrow;

  const v16bf* WsP = (const v16bf*)WselfPk;    // packed: [(tile*4 + kkidx)*32 + lane]
  const v16bf* WnP = (const v16bf*)WneighPk;

  v8f acc0 = {0.f, 0.f, 0.f, 0.f, 0.f, 0.f, 0.f, 0.f};
  v8f acc1 = {0.f, 0.f, 0.f, 0.f, 0.f, 0.f, 0.f, 0.f};

#pragma unroll
  for (int kkidx = 0; kkidx < 4; ++kkidx) {
    const int kk = kkidx * 32;
    v16bf aS0, aN0, aS1, aN1;
#pragma unroll
    for (int e = 0; e < 16; ++e) {
      int ka = kk + ((e < 8) ? 0 : 16) + kbA + (e & 7);
      aS0[e] = As[mrow][ka];
      aN0[e] = An[mrow][ka];
      aS1[e] = As[mrow + 16][ka];
      aN1[e] = An[mrow + 16][ka];
    }
    v16bf bS = WsP[(wave * 4 + kkidx) * 32 + lane];   // one 32B coalesced load
    v16bf bN = WnP[(wave * 4 + kkidx) * 32 + lane];
    acc0 = __builtin_amdgcn_wmma_f32_16x16x32_bf16(false, aS0, false, bS, (short)0, acc0, false, false);
    acc0 = __builtin_amdgcn_wmma_f32_16x16x32_bf16(false, aN0, false, bN, (short)0, acc0, false, false);
    acc1 = __builtin_amdgcn_wmma_f32_16x16x32_bf16(false, aS1, false, bS, (short)0, acc1, false, false);
    acc1 = __builtin_amdgcn_wmma_f32_16x16x32_bf16(false, aN1, false, bN, (short)0, acc1, false, false);
  }

  const float bv = bias[wcol];
  const int r0 = (lane < 16) ? 0 : 8;
#pragma unroll
  for (int i = 0; i < 8; ++i) {
    float v0 = acc0[i] + bv;
    float v1 = acc1[i] + bv;
    hout[(size_t)(rowBase + r0 + i) * DGR + wcol]      = fmaxf(v0, 0.0f);
    hout[(size_t)(rowBase + 16 + r0 + i) * DGR + wcol] = fmaxf(v1, 0.0f);
  }
}

// ---------------------------------------------------------------- graph mean pool
__global__ void gnn_pool_scatter(const int* __restrict__ gid, const float* __restrict__ h,
                                 float* __restrict__ gsum, float* __restrict__ gcnt, int N) {
  int tid = blockIdx.x * blockDim.x + threadIdx.x;   // wave per node
  if (tid >= N * 32) return;
  int node = tid >> 5, lane = tid & 31;
  int g = gid[node];
  float4 v = ((const float4*)(h + (size_t)node * DGR))[lane];
  float* gp = gsum + (size_t)g * DGR + lane * 4;
  atomicAdd(gp + 0, v.x); atomicAdd(gp + 1, v.y);
  atomicAdd(gp + 2, v.z); atomicAdd(gp + 3, v.w);
  if (lane == 0) atomicAdd(gcnt + g, 1.0f);
}

__global__ void gnn_pool_div(const float* __restrict__ gsum, const float* __restrict__ gcnt,
                             float* __restrict__ hg, int n) {
  int i = blockIdx.x * blockDim.x + threadIdx.x;
  if (i < n) hg[i] = gsum[i] / fmaxf(gcnt[i >> 7], 1.0f);
}

// ---------------------------------------------------------------- WMMA MLP layer
// C[M,Nw] = act(A[M,K] @ W[K,Nw] + b); block = 16 rows x 128 cols, 8 waves;
// W pre-packed in fragment order.
__global__ __launch_bounds__(256)
void gnn_gemm_bias_act(const float* __restrict__ A, const bf16_t* __restrict__ Wpk,
                       const float* __restrict__ bias, float* __restrict__ C,
                       int M, int K, int Nw, int relu) {
  __shared__ bf16_t As[16][264];
  const int rowBase = blockIdx.x * 16;
  const int t = threadIdx.x;

  {  // stage 16 x K tile of A, K/16 elems per thread
    int per = K >> 4;
    int idx = t * per;
    int r = idx / K, c = idx - r * K;
    const float* ap = A + (size_t)(rowBase + r) * K + c;
    for (int i = 0; i < per; ++i) As[r][c + i] = (bf16_t)ap[i];
  }
  __syncthreads();

  const int wave = t >> 5;
  const int lane = t & 31;
  const int tile = blockIdx.y * 8 + wave;   // global 16-col tile index
  const int mrow = lane & 15;
  const int kbA  = (lane >> 4) * 8;
  const int wcol = tile * 16 + mrow;
  const int nkk  = K >> 5;

  const v16bf* WP = (const v16bf*)Wpk;

  v8f acc = {0.f, 0.f, 0.f, 0.f, 0.f, 0.f, 0.f, 0.f};

  for (int kkidx = 0; kkidx < nkk; ++kkidx) {
    const int kk = kkidx * 32;
    v16bf a;
#pragma unroll
    for (int e = 0; e < 16; ++e) {
      int ka = kk + ((e < 8) ? 0 : 16) + kbA + (e & 7);
      a[e] = As[mrow][ka];
    }
    v16bf b = WP[(tile * nkk + kkidx) * 32 + lane];
    acc = __builtin_amdgcn_wmma_f32_16x16x32_bf16(false, a, false, b, (short)0, acc, false, false);
  }

  const float bv = bias[wcol];
  const int r0 = (lane < 16) ? 0 : 8;
#pragma unroll
  for (int i = 0; i < 8; ++i) {
    float v = acc[i] + bv;
    if (relu) v = fmaxf(v, 0.0f);
    C[(size_t)(rowBase + r0 + i) * Nw + wcol] = v;
  }
}

// ---------------------------------------------------------------- prediction head
__global__ void gnn_pred_head(const float* __restrict__ h4, const float* __restrict__ pw,
                              const float* __restrict__ pb, float* __restrict__ out) {
  int r = blockIdx.x * blockDim.x + threadIdx.x;
  if (r >= NGRAPH) return;
  float y0 = pb[0], y1 = pb[1];
  const float* hp = h4 + (size_t)r * HID;
  for (int k = 0; k < HID; ++k) {
    float hv = hp[k];
    y0 += hv * pw[k * 2 + 0];
    y1 += hv * pw[k * 2 + 1];
  }
  float m = fmaxf(y0, y1);
  float lse = m + __logf(__expf(y0 - m) + __expf(y1 - m));
  out[r * 2 + 0] = y0 - lse;
  out[r * 2 + 1] = y1 - lse;
}

// ---------------------------------------------------------------- launcher
extern "C" void kernel_launch(void* const* d_in, const int* in_sizes, int n_in,
                              void* d_out, int out_size, void* d_ws, size_t ws_size,
                              hipStream_t stream) {
  const int*   feat = (const int*)d_in[0];
  const int*   src  = (const int*)d_in[1];
  const int*   dst  = (const int*)d_in[2];
  const int*   gid  = (const int*)d_in[3];
  const float* emb  = (const float*)d_in[4];
  const float* ws0  = (const float*)d_in[5];
  const float* wn0  = (const float*)d_in[6];
  const float* b0   = (const float*)d_in[7];
  const float* ws1  = (const float*)d_in[8];
  const float* wn1  = (const float*)d_in[9];
  const float* b1   = (const float*)d_in[10];
  const float* e1w  = (const float*)d_in[11];
  const float* e1b  = (const float*)d_in[12];
  const float* e2w  = (const float*)d_in[13];
  const float* e2b  = (const float*)d_in[14];
  const float* e3w  = (const float*)d_in[15];
  const float* e3b  = (const float*)d_in[16];
  const float* e4w  = (const float*)d_in[17];
  const float* e4b  = (const float*)d_in[18];
  const float* pw   = (const float*)d_in[19];
  const float* pb   = (const float*)d_in[20];

  const int N = in_sizes[0];
  const int E = in_sizes[1];

  // ---- carve workspace
  char* wsp = (char*)d_ws;
  auto carve = [&](size_t bytes) -> char* {
    char* p = wsp;
    wsp += (bytes + 255) & ~(size_t)255;
    return p;
  };
  float*  hA   = (float*)carve((size_t)N * DGR * 4);
  float*  hB   = (float*)carve((size_t)N * DGR * 4);
  float*  agg  = (float*)carve((size_t)N * DGR * 4);
  float*  deg  = (float*)carve((size_t)N * 4);
  float*  gsum = (float*)carve((size_t)NGRAPH * DGR * 4);
  float*  gcnt = (float*)carve((size_t)NGRAPH * 4);
  float*  hg   = (float*)carve((size_t)NGRAPH * DGR * 4);
  float*  a2   = (float*)carve((size_t)NGRAPH * HID * 4);
  float*  a3   = (float*)carve((size_t)NGRAPH * HID * 4);
  float*  a4   = (float*)carve((size_t)NGRAPH * HID * 4);
  bf16_t* cws0 = (bf16_t*)carve((size_t)DGR * DGR * 2);
  bf16_t* cwn0 = (bf16_t*)carve((size_t)DGR * DGR * 2);
  bf16_t* cws1 = (bf16_t*)carve((size_t)DGR * DGR * 2);
  bf16_t* cwn1 = (bf16_t*)carve((size_t)DGR * DGR * 2);
  bf16_t* cw1  = (bf16_t*)carve((size_t)DGR * HID * 2);
  bf16_t* cw2  = (bf16_t*)carve((size_t)HID * HID * 2);
  bf16_t* cw3  = (bf16_t*)carve((size_t)HID * HID * 2);
  bf16_t* cw4  = (bf16_t*)carve((size_t)HID * HID * 2);

  float* out    = (float*)d_out;
  float* hidden = out + NGRAPH * NOUT;   // second tuple output lives in d_out

  // ---- pack weights into bf16 WMMA-fragment order (one pass, L2-resident after)
  gnn_pack_w<<<(DGR * DGR + 255) / 256, 256, 0, stream>>>(ws0, cws0, DGR, DGR);
  gnn_pack_w<<<(DGR * DGR + 255) / 256, 256, 0, stream>>>(wn0, cwn0, DGR, DGR);
  gnn_pack_w<<<(DGR * DGR + 255) / 256, 256, 0, stream>>>(ws1, cws1, DGR, DGR);
  gnn_pack_w<<<(DGR * DGR + 255) / 256, 256, 0, stream>>>(wn1, cwn1, DGR, DGR);
  gnn_pack_w<<<(DGR * HID + 255) / 256, 256, 0, stream>>>(e1w, cw1, DGR, HID);
  gnn_pack_w<<<(HID * HID + 255) / 256, 256, 0, stream>>>(e2w, cw2, HID, HID);
  gnn_pack_w<<<(HID * HID + 255) / 256, 256, 0, stream>>>(e3w, cw3, HID, HID);
  gnn_pack_w<<<(HID * HID + 255) / 256, 256, 0, stream>>>(e4w, cw4, HID, HID);

  // ---- embedding gather
  gnn_gather_embed<<<(N * 32 + 255) / 256, 256, 0, stream>>>(feat, emb, hA, N);

  // ---- layer 0
  gnn_zero_f32<<<4096, 256, 0, stream>>>(agg, (size_t)N * DGR);
  gnn_zero_f32<<<512, 256, 0, stream>>>(deg, (size_t)N);
  gnn_deg_count<<<(E + 255) / 256, 256, 0, stream>>>(dst, deg, E);
  gnn_edge_scatter<<<(E * 32 + 255) / 256, 256, 0, stream>>>(src, dst, hA, agg, E);
  gnn_sage_update<<<N / 32, 256, 0, stream>>>(hA, agg, deg, cws0, cwn0, b0, hB, N);

  // ---- layer 1 (deg is identical, reuse)
  gnn_zero_f32<<<4096, 256, 0, stream>>>(agg, (size_t)N * DGR);
  gnn_edge_scatter<<<(E * 32 + 255) / 256, 256, 0, stream>>>(src, dst, hB, agg, E);
  gnn_sage_update<<<N / 32, 256, 0, stream>>>(hB, agg, deg, cws1, cwn1, b1, hA, N);

  // ---- per-graph mean pool
  gnn_zero_f32<<<64, 256, 0, stream>>>(gsum, (size_t)NGRAPH * DGR);
  gnn_zero_f32<<<1, 256, 0, stream>>>(gcnt, (size_t)NGRAPH);
  gnn_pool_scatter<<<(N * 32 + 255) / 256, 256, 0, stream>>>(gid, hA, gsum, gcnt, N);
  gnn_pool_div<<<(NGRAPH * DGR + 255) / 256, 256, 0, stream>>>(gsum, gcnt, hg, NGRAPH * DGR);

  // ---- MLP (WMMA), hidden written straight into d_out tuple slot
  gnn_gemm_bias_act<<<dim3(NGRAPH / 16, HID / 128), 256, 0, stream>>>(hg, cw1, e1b, hidden, NGRAPH, DGR, HID, 1);
  gnn_gemm_bias_act<<<dim3(NGRAPH / 16, HID / 128), 256, 0, stream>>>(hidden, cw2, e2b, a2, NGRAPH, HID, HID, 1);
  gnn_gemm_bias_act<<<dim3(NGRAPH / 16, HID / 128), 256, 0, stream>>>(a2, cw3, e3b, a3, NGRAPH, HID, HID, 1);
  gnn_gemm_bias_act<<<dim3(NGRAPH / 16, HID / 128), 256, 0, stream>>>(a3, cw4, e4b, a4, NGRAPH, HID, HID, 1);

  // ---- prediction + log_softmax
  gnn_pred_head<<<1, 256, 0, stream>>>(a4, pw, pb, out);
}
